// LearnableMidiSynth_46763603918959
// MI455X (gfx1250) — matrix operand order repaired and checked by reading, not compiled, not run on hardware
//
#include <hip/hip_runtime.h>
#include <hip/hip_bf16.h>

typedef __attribute__((ext_vector_type(16))) _Float16 v16h;
typedef __attribute__((ext_vector_type(8)))  float    v8f;

#define NN   2048
#define LL   2646000
#define SS   16384
#define WINW 512
#define KH   16

// ---------------------------------------------------------------------------
// Prep kernel: one wave (32 lanes) per block, 16 notes per block.
// - scalar time-MLP (1->32->2) and delay-MLP (1->64->1) per lane (lanes 0..15)
// - WMMA f16 16x16x32 for feat(4->pad32)->64 (4 column tiles) and 64->16
//   (2 K-chunks accumulated), per CDNA5 VGPR layouts, staged through LDS.
// Outputs per note: hamps[16] (float) and meta {out_len, out_len2, new_start}.
// ---------------------------------------------------------------------------
__global__ __launch_bounds__(32)
void midi_prep_kernel(const float* __restrict__ freq_rad,
                      const float* __restrict__ velocity,
                      const int*   __restrict__ start_sample,
                      const int*   __restrict__ end_sample,
                      const float* __restrict__ tW1, const float* __restrict__ tb1,
                      const float* __restrict__ tW2, const float* __restrict__ tb2,
                      const float* __restrict__ hW1, const float* __restrict__ hb1,
                      const float* __restrict__ hW2, const float* __restrict__ hb2,
                      const float* __restrict__ dW1, const float* __restrict__ db1,
                      const float* __restrict__ dW2, const float* __restrict__ db2,
                      float* __restrict__ ws_hamps,  // [NN][16]
                      int*   __restrict__ ws_meta)   // [NN][3]
{
    __shared__ _Float16 sW1[32][64];   // hW1 padded K: 4 -> 32
    __shared__ _Float16 sW2[64][16];   // hW2
    __shared__ _Float16 sFeat[16][32]; // 16 notes x padded-K feat
    __shared__ _Float16 sHH[16][64];   // hidden activations

    const int tid  = threadIdx.x;       // 0..31
    const int m    = tid & 15;
    const int khi  = tid >> 4;          // 0 or 1 (lane half)
    const int base = blockIdx.x * 16;   // first note of this block

    // Stage weights into LDS as f16 (zero-pad hW1 rows 4..31).
    for (int i = tid; i < 32 * 64; i += 32) {
        int k = i >> 6, c = i & 63;
        sW1[k][c] = (k < 4) ? (_Float16)hW1[k * 64 + c] : (_Float16)0.0f;
    }
    for (int i = tid; i < 64 * 16; i += 32) {
        int k = i >> 4, c = i & 15;
        sW2[k][c] = (_Float16)hW2[k * 16 + c];
    }

    // Per-note scalar MLPs (lanes 0..15, one note each).
    if (tid < 16) {
        const int note = base + tid;
        const int st   = start_sample[note];
        const int en   = end_sample[note];
        const float f  = freq_rad[note];
        const float nl = (float)st / (float)LL;
        const int out_len = ((en < LL) ? en : LL) - st;

        // time MLP: 1 -> 32 -> 2
        float tl0 = tb2[0], tl1 = tb2[1];
        #pragma unroll 4
        for (int j = 0; j < 32; ++j) {
            float h = nl * tW1[j] + tb1[j];
            h = (h > 0.0f) ? h : 0.0f;
            tl0 += h * tW2[2 * j + 0];
            tl1 += h * tW2[2 * j + 1];
        }

        // delay MLP: 1 -> 64 -> 1
        float dacc = db2[0];
        #pragma unroll 4
        for (int j = 0; j < 64; ++j) {
            float h = nl * dW1[j] + db1[j];
            h = (h > 0.0f) ? h : 0.0f;
            dacc += h * dW2[j];
        }
        float d = dacc * 0.01f;
        d = (d > 0.01f) ? 0.01f : ((d < -0.01f) ? -0.01f : d);
        d += nl;
        const int new_start = (int)(d * (float)LL);  // trunc toward zero
        const int out_len2  = (new_start + out_len >= LL) ? (LL - new_start) : out_len;

        ws_meta[note * 3 + 0] = out_len;
        ws_meta[note * 3 + 1] = out_len2;
        ws_meta[note * 3 + 2] = new_start;

        // feat = [freq, tl0, tl1, vel/127], zero-padded to K=32
        sFeat[tid][0] = (_Float16)f;
        sFeat[tid][1] = (_Float16)tl0;
        sFeat[tid][2] = (_Float16)tl1;
        sFeat[tid][3] = (_Float16)(velocity[note] * (1.0f / 127.0f));
        #pragma unroll
        for (int k = 4; k < 32; ++k) sFeat[tid][k] = (_Float16)0.0f;
    }
    __syncthreads();

    // ---- Layer 1: (16x32 feat) x (32x16 W1-tile) -> 16x16, 4 column tiles
    // A fragment (16-bit A 16x32 layout): lane<16: M=lane, K in {0..7,16..23};
    // lane>=16: M=lane-16, K in {8..15,24..31}.
    v16h a;
    #pragma unroll
    for (int j = 0; j < 16; ++j) {
        int K = ((j >> 3) << 4) + (j & 7) + khi * 8;
        a[j] = sFeat[m][K];
    }

    #pragma unroll
    for (int c = 0; c < 4; ++c) {
        const int col = c * 16 + m;
        v16h b;
        #pragma unroll
        for (int j = 0; j < 16; ++j) b[j] = sW1[((j >> 3) << 4) + (j & 7) + khi * 8][col];  // lanes0-15:K{0..7,16..23}, lanes16-31:K{8..15,24..31}
        v8f z = {};
        v8f acc = __builtin_amdgcn_wmma_f32_16x16x32_f16(
            false, a, false, b, (short)0, z, false, false);
        // C/D layout: VGPR j -> row (j + 8*khi), col m (within tile) -> global col
        const float bias = hb1[col];
        #pragma unroll
        for (int j = 0; j < 8; ++j) {
            float v = acc[j] + bias;
            sHH[j + 8 * khi][col] = (_Float16)((v > 0.0f) ? v : 0.0f);
        }
    }
    __syncthreads();

    // ---- Layer 2: (16x64 hh) x (64x16 W2) -> 16x16 hamps, 2 K-chunks accumulated
    v8f acc2 = {};
    #pragma unroll
    for (int q = 0; q < 2; ++q) {
        v16h a2, b2;
        #pragma unroll
        for (int j = 0; j < 16; ++j) {
            int K = q * 32 + ((j >> 3) << 4) + (j & 7) + khi * 8;
            a2[j] = sHH[m][K];
        }
        #pragma unroll
        for (int j = 0; j < 16; ++j) b2[j] = sW2[q * 32 + ((j >> 3) << 4) + (j & 7) + khi * 8][m];
        acc2 = __builtin_amdgcn_wmma_f32_16x16x32_f16(
            false, a2, false, b2, (short)0, acc2, false, false);
    }
    #pragma unroll
    for (int j = 0; j < 8; ++j) {
        const int row  = j + 8 * khi;           // note index within tile
        const int note = base + row;
        ws_hamps[note * KH + m] = acc2[j] + hb2[m];
    }
}

// ---------------------------------------------------------------------------
// Synth kernel: grid (SS/256, NN), block 256. One sample per thread.
// sin(f*(k+1)*n + t) via complex rotation recurrence: 2 transcendentals per
// sample, ~5 VALU ops per harmonic, then fade window + global f32 atomicAdd
// (output fits in L2, atomics resolve there).
// ---------------------------------------------------------------------------
__global__ __launch_bounds__(256)
void midi_synth_kernel(const float* __restrict__ freq_rad,
                       const float* __restrict__ t_ptr,
                       const float* __restrict__ ws_hamps,
                       const int*   __restrict__ ws_meta,
                       float* __restrict__ out)
{
    __shared__ float sh[KH];
    __shared__ int   sm[3];

    const int note = blockIdx.y;
    if (threadIdx.x < KH) sh[threadIdx.x] = ws_hamps[note * KH + threadIdx.x];
    if (threadIdx.x < 3)  sm[threadIdx.x] = ws_meta[note * 3 + threadIdx.x];
    __syncthreads();

    const int out_len   = sm[0];
    const int out_len2  = sm[1];
    const int new_start = sm[2];

    const int n = blockIdx.x * blockDim.x + threadIdx.x;
    if (n >= out_len2) return;                 // beyond valid region: adds 0
    const int idx = new_start + n;
    if (idx < 0 || idx >= LL) return;          // reference masks these to 0

    const float f = freq_rad[note];
    const float t = t_ptr[0];

    const float beta = f * (float)n;
    const float sb = sinf(beta), cb = cosf(beta);
    const float st = sinf(t),    ct = cosf(t);

    // z = e^{i(beta + t)}; harmonic k+1 phase = (k+1)*beta + t
    float zr = ct * cb - st * sb;
    float zi = st * cb + ct * sb;
    float acc = sh[0] * zi;
    #pragma unroll
    for (int k = 1; k < KH; ++k) {
        float nr = zr * cb - zi * sb;
        zi       = zr * sb + zi * cb;
        zr       = nr;
        acc += sh[k] * zi;
    }

    // fade window over last WINW samples before out_len (n < out_len guaranteed
    // since out_len2 <= out_len)
    float fac = 1.0f;
    const int wstart = out_len - WINW;
    if (n >= wstart)
        fac = 1.0f - (float)(n - wstart) * (1.0f / (float)(WINW - 1));

    atomicAdd(&out[idx], acc * fac);
}

extern "C" void kernel_launch(void* const* d_in, const int* in_sizes, int n_in,
                              void* d_out, int out_size, void* d_ws, size_t ws_size,
                              hipStream_t stream) {
    const float* freq_rad     = (const float*)d_in[0];
    const float* velocity     = (const float*)d_in[1];
    const int*   start_sample = (const int*)d_in[2];
    const int*   end_sample   = (const int*)d_in[3];
    const float* output_in    = (const float*)d_in[4];
    const float* t_ptr        = (const float*)d_in[5];
    const float* tW1 = (const float*)d_in[6];
    const float* tb1 = (const float*)d_in[7];
    const float* tW2 = (const float*)d_in[8];
    const float* tb2 = (const float*)d_in[9];
    const float* hW1 = (const float*)d_in[10];
    const float* hb1 = (const float*)d_in[11];
    const float* hW2 = (const float*)d_in[12];
    const float* hb2 = (const float*)d_in[13];
    const float* dW1 = (const float*)d_in[14];
    const float* db1 = (const float*)d_in[15];
    const float* dW2 = (const float*)d_in[16];
    const float* db2 = (const float*)d_in[17];

    float* out = (float*)d_out;

    // Workspace: hamps [NN*16 floats], meta [NN*3 ints]
    float* ws_hamps = (float*)d_ws;
    int*   ws_meta  = (int*)((char*)d_ws + (size_t)NN * KH * sizeof(float));

    // output starts as the provided (zero) input buffer
    hipMemcpyAsync(out, output_in, (size_t)out_size * sizeof(float),
                   hipMemcpyDeviceToDevice, stream);

    // Per-note MLPs (WMMA path): 2048 notes / 16 per block = 128 blocks x 1 wave
    midi_prep_kernel<<<NN / 16, 32, 0, stream>>>(
        freq_rad, velocity, start_sample, end_sample,
        tW1, tb1, tW2, tb2, hW1, hb1, hW2, hb2, dW1, db1, dW2, db2,
        ws_hamps, ws_meta);

    // Synthesis + scatter-add
    dim3 grid(SS / 256, NN);
    midi_synth_kernel<<<grid, 256, 0, stream>>>(
        freq_rad, t_ptr, ws_hamps, ws_meta, out);
}
